// PointNet2_seg_23278722744327
// MI455X (gfx1250) — compile-verified, matrix operand order checked by reading, not run on hardware
//
#include <hip/hip_runtime.h>
#include <hip/hip_bf16.h>

// ---------------------------------------------------------------------------
// Problem constants (match reference)
// ---------------------------------------------------------------------------
#define BB   2
#define NN   4096
#define II   16
#define OO   128
#define KK   64
#define NPTS (BB * NN)          // 8192
#define R2   0.015625f          // 0.125^2
#define BNEPS 1e-5f
#define FBIG 3.0e38f

typedef __attribute__((ext_vector_type(16))) _Float16 v16h;
typedef __attribute__((ext_vector_type(8)))  _Float16 v8h;
typedef __attribute__((ext_vector_type(8)))  float    v8f;

// ---------------------------------------------------------------------------
// WMMA fragment loaders (wave32 layouts, ISA 7.12.2), vectorized as two
// 128-bit LDS loads per fragment (ds_load_b128).
//   A: 16x32 f16 tile, row-major; lane&15 = M row, lane>>4 = K sub-half.
//      Lane needs k = k0+lh*8..+7 and k0+16+lh*8..+7  -> two contiguous v8h.
//   B: stored TRANSPOSED in LDS (n-major, 32 k per column) so the same two
//      contiguous v8h loads apply with lane&15 = N column.
//   C/D: lane&15 = N, vgpr r = M row (+8 for lanes 16..31).
// ---------------------------------------------------------------------------
__device__ __forceinline__ v16h fuse16(v8h lo, v8h hi) {
  return __builtin_shufflevector(lo, hi, 0, 1, 2, 3, 4, 5, 6, 7,
                                 8, 9, 10, 11, 12, 13, 14, 15);
}

__device__ __forceinline__ v16h load_frag_a(const _Float16* src, int lda,
                                            int row0, int k0, int lane) {
  int lh = (lane >> 4) & 1;
  int m  = lane & 15;
  const _Float16* p = src + (size_t)(row0 + m) * lda + k0 + lh * 8;
  v8h lo = *(const v8h*)(p);
  v8h hi = *(const v8h*)(p + 16);
  return fuse16(lo, hi);
}

// srcT layout: [128][32] halves, element (k,n) at srcT[n*32 + k]
__device__ __forceinline__ v16h load_frag_b_t(const _Float16* srcT,
                                              int n0, int lane) {
  int lh = (lane >> 4) & 1;
  int n  = lane & 15;
  const _Float16* p = srcT + (size_t)(n0 + n) * 32 + lh * 8;
  v8h lo = *(const v8h*)(p);
  v8h hi = *(const v8h*)(p + 16);
  return fuse16(lo, hi);
}

// Chunked-K GEMM: A (64 rows staged in LDS, f16) x W (global f32, streamed
// through a transposed 32x128 f16 LDS chunk).  Each wave accumulates a 16x128
// strip into acc[8].  All 128 threads must call this uniformly.
// B fragments for all 8 N-tiles are batch-loaded (one ds clause, one wait)
// before the 8 back-to-back WMMAs so LDS latency is amortized.
template <int KTILES>
__device__ __forceinline__ void gemm_acc(const _Float16* Amat, int lda, int arow0,
                                         const float* Wg, int cinRows,
                                         _Float16* wChunkT, v8f* acc,
                                         int tid, int lane) {
#pragma unroll 1
  for (int kt = 0; kt < KTILES; ++kt) {
    __syncthreads();
    {
      int nn = tid;                       // 128 threads -> one column each
#pragma unroll
      for (int kb = 0; kb < 32; kb += 8) {
        v8h pk;
#pragma unroll
        for (int j = 0; j < 8; ++j) {
          int kr = kt * 32 + kb + j;
          pk[j] = (_Float16)((kr < cinRows) ? Wg[(size_t)kr * 128 + nn] : 0.0f);
        }
        *(v8h*)(wChunkT + (size_t)nn * 32 + kb) = pk;
      }
    }
    __syncthreads();
    v16h a = load_frag_a(Amat, lda, arow0, kt * 32, lane);
    v16h bfr[8];
#pragma unroll
    for (int nt = 0; nt < 8; ++nt) bfr[nt] = load_frag_b_t(wChunkT, nt * 16, lane);
#pragma unroll
    for (int nt = 0; nt < 8; ++nt) {
      acc[nt] = __builtin_amdgcn_wmma_f32_16x16x32_f16(
          false, a, false, bfr[nt], (short)0, acc[nt], false, false);
    }
  }
}

__device__ __forceinline__ void add_bias(v8f* acc, const float* b, int nl) {
#pragma unroll
  for (int nt = 0; nt < 8; ++nt) {
    float bias = b[nt * 16 + nl];
#pragma unroll
    for (int r = 0; r < 8; ++r) acc[nt][r] += bias;
  }
}

// ---------------------------------------------------------------------------
// Ball-query kNN: one workgroup per query point; d2 table in LDS, K argmin
// sweeps.  idx stores GLOBAL neighbor row, or -1 if outside radius.
// ---------------------------------------------------------------------------
__global__ __launch_bounds__(256) void knn_kernel(const float* __restrict__ pos,
                                                  int* __restrict__ idxOut,
                                                  float* __restrict__ cnt) {
  __shared__ float d2s[NN];
  __shared__ float rv[256];
  __shared__ int   ri[256];
  int tid  = threadIdx.x;
  int gid  = blockIdx.x;          // 0..NPTS-1
  int base = (gid / NN) * NN;     // batch start
  float px = pos[gid * 3 + 0], py = pos[gid * 3 + 1], pz = pos[gid * 3 + 2];
  for (int j = tid; j < NN; j += 256) {
    float dx = pos[(base + j) * 3 + 0] - px;
    float dy = pos[(base + j) * 3 + 1] - py;
    float dz = pos[(base + j) * 3 + 2] - pz;
    float d  = dx * dx + dy * dy + dz * dz;
    d2s[j]   = (d <= R2) ? d : FBIG;
  }
  __syncthreads();
  float nv = 0.0f;
  for (int s = 0; s < KK; ++s) {
    float best = FBIG;
    int   bi   = -1;
    for (int j = tid; j < NN; j += 256) {
      float v = d2s[j];
      if (v < best) { best = v; bi = j; }
    }
    rv[tid] = best; ri[tid] = bi;
    __syncthreads();
    for (int o = 128; o > 0; o >>= 1) {
      if (tid < o && rv[tid + o] < rv[tid]) { rv[tid] = rv[tid + o]; ri[tid] = ri[tid + o]; }
      __syncthreads();
    }
    float sv = rv[0];
    int   sel = ri[0];
    bool  found = (sv < FBIG) && (sel >= 0);
    if (tid == 0) {
      idxOut[(size_t)gid * KK + s] = found ? (base + sel) : -1;
      if (found) { d2s[sel] = FBIG; nv += 1.0f; }
    }
    __syncthreads();
  }
  if (tid == 0) atomicAdd(cnt, nv);
}

// ---------------------------------------------------------------------------
// Small helpers
// ---------------------------------------------------------------------------
__global__ void zero_f32(float* p, int n) {
  int i = blockIdx.x * blockDim.x + threadIdx.x;
  if (i < n) p[i] = 0.0f;
}

__global__ __launch_bounds__(128) void bn_finalize(const float* __restrict__ sum,
                                                   const float* __restrict__ sq,
                                                   const float* __restrict__ cnt,
                                                   const float* __restrict__ g,
                                                   const float* __restrict__ be,
                                                   float* __restrict__ scale,
                                                   float* __restrict__ shift) {
  int c = threadIdx.x;
  float n    = fmaxf(cnt[0], 1.0f);
  float mean = sum[c] / n;
  float var  = sq[c] / n - mean * mean;
  float sc   = rsqrtf(var + BNEPS) * g[c];
  scale[c] = sc;
  shift[c] = be[c] - mean * sc;
}

// ---------------------------------------------------------------------------
// Fused edge-MLP kernel.  One workgroup (128 thr = 4 waves) per destination
// node; its 64 edges form the GEMM M dimension.
//   STAGE 0: e@W1+b1, accumulate BN1 stats (masked), exit.
//   STAGE 1: + BN1/leaky, h@W2+b2, accumulate BN2 stats, exit.
//   STAGE 2: + BN2/leaky, masked max over K -> agg[node][128].
// ---------------------------------------------------------------------------
template <int STAGE, int CIN>
__global__ __launch_bounds__(128) void edge_mlp(
    const float* __restrict__ pos, const float* __restrict__ xin,
    const int* __restrict__ idx,
    const float* __restrict__ W1, const float* __restrict__ b1,
    const float* __restrict__ W2, const float* __restrict__ b2,
    const float* __restrict__ scale1, const float* __restrict__ shift1,
    const float* __restrict__ scale2, const float* __restrict__ shift2,
    float* __restrict__ gSum1, float* __restrict__ gSq1,
    float* __restrict__ gSum2, float* __restrict__ gSq2,
    float* __restrict__ agg) {
  constexpr int KE1 = (CIN + 3 + 31) & ~31;   // 32 (layer 0) or 160
  constexpr int KT1 = KE1 / 32;

  __shared__ alignas(16) _Float16 eT[64 * KE1];       // edge features, f16
  __shared__ alignas(16) _Float16 hT[64 * 128];       // hidden post BN1/leaky
  __shared__ alignas(16) _Float16 wChunkT[128 * 32];  // streamed W (transposed)
  __shared__ float red[2][128];                       // stat reduction
  __shared__ float wmaxs[4][128];                     // per-wave channel max
  __shared__ int   vmask[64];
  __shared__ int   jrow[64];

  int tid  = threadIdx.x;
  int lane = tid & 31;
  int wv   = tid >> 5;
  int lh   = lane >> 4;
  int nl   = lane & 15;
  int gid  = blockIdx.x;   // node id

  if (tid < 64) {
    int j = idx[(size_t)gid * KK + tid];
    vmask[tid] = (j >= 0) ? 1 : 0;
    jrow[tid]  = (j >= 0) ? j : gid;
  }
  __syncthreads();

  float p0 = pos[gid * 3 + 0], p1 = pos[gid * 3 + 1], p2 = pos[gid * 3 + 2];
  for (int t = tid; t < 64 * KE1; t += 128) {
    int r = t / KE1, c = t % KE1;
    int j = jrow[r];
    bool val = vmask[r] != 0;
    float v;
    if (c < CIN)            v = val ? xin[(size_t)j * CIN + c] : 0.0f;
    else if (c < CIN + 3) {
      int k = c - CIN;
      float pi = (k == 0) ? p0 : (k == 1 ? p1 : p2);
      v = val ? (pos[j * 3 + k] - pi) : 0.0f;
    } else v = 0.0f;
    eT[t] = (_Float16)v;
  }

  // ---- GEMM1: e @ W1 (+b1) --------------------------------------------
  v8f acc[8];
#pragma unroll
  for (int nt = 0; nt < 8; ++nt) acc[nt] = (v8f){0.f,0.f,0.f,0.f,0.f,0.f,0.f,0.f};
  gemm_acc<KT1>(eT, KE1, wv * 16, W1, CIN + 3, wChunkT, acc, tid, lane);
  add_bias(acc, b1, nl);

  if constexpr (STAGE == 0) {
    if (tid < 128) { red[0][tid] = 0.0f; red[1][tid] = 0.0f; }
    __syncthreads();
#pragma unroll
    for (int nt = 0; nt < 8; ++nt) {
      float s = 0.0f, q = 0.0f;
#pragma unroll
      for (int r = 0; r < 8; ++r) {
        int row = wv * 16 + r + 8 * lh;
        if (vmask[row]) { float v = acc[nt][r]; s += v; q += v * v; }
      }
      atomicAdd(&red[0][nt * 16 + nl], s);
      atomicAdd(&red[1][nt * 16 + nl], q);
    }
    __syncthreads();
    if (tid < 128) { atomicAdd(&gSum1[tid], red[0][tid]); atomicAdd(&gSq1[tid], red[1][tid]); }
    return;
  } else {
    // ---- BN1 + leaky -> hT ------------------------------------------
#pragma unroll
    for (int nt = 0; nt < 8; ++nt) {
      int n = nt * 16 + nl;
      float sc = scale1[n], sh = shift1[n];
#pragma unroll
      for (int r = 0; r < 8; ++r) {
        int row = wv * 16 + r + 8 * lh;
        float v = acc[nt][r] * sc + sh;
        v = (v > 0.0f) ? v : 0.2f * v;
        if (!vmask[row]) v = 0.0f;
        hT[(size_t)row * 128 + n] = (_Float16)v;
      }
    }

    // ---- GEMM2: h @ W2 (+b2) ----------------------------------------
    v8f acc2[8];
#pragma unroll
    for (int nt = 0; nt < 8; ++nt) acc2[nt] = (v8f){0.f,0.f,0.f,0.f,0.f,0.f,0.f,0.f};
    gemm_acc<4>(hT, 128, wv * 16, W2, 128, wChunkT, acc2, tid, lane);
    add_bias(acc2, b2, nl);

    if constexpr (STAGE == 1) {
      if (tid < 128) { red[0][tid] = 0.0f; red[1][tid] = 0.0f; }
      __syncthreads();
#pragma unroll
      for (int nt = 0; nt < 8; ++nt) {
        float s = 0.0f, q = 0.0f;
#pragma unroll
        for (int r = 0; r < 8; ++r) {
          int row = wv * 16 + r + 8 * lh;
          if (vmask[row]) { float v = acc2[nt][r]; s += v; q += v * v; }
        }
        atomicAdd(&red[0][nt * 16 + nl], s);
        atomicAdd(&red[1][nt * 16 + nl], q);
      }
      __syncthreads();
      if (tid < 128) { atomicAdd(&gSum2[tid], red[0][tid]); atomicAdd(&gSq2[tid], red[1][tid]); }
      return;
    } else {
      // ---- STAGE 2: BN2 + leaky + masked max over the 64 edges ------
#pragma unroll
      for (int nt = 0; nt < 8; ++nt) {
        int n = nt * 16 + nl;
        float sc = scale2[n], sh = shift2[n];
        float mx = -FBIG;
#pragma unroll
        for (int r = 0; r < 8; ++r) {
          int row = wv * 16 + r + 8 * lh;
          float v = acc2[nt][r] * sc + sh;
          v = (v > 0.0f) ? v : 0.2f * v;
          if (vmask[row]) mx = fmaxf(mx, v);
        }
        mx = fmaxf(mx, __shfl_xor(mx, 16, 32));
        if (lane < 16) wmaxs[wv][n] = mx;
      }
      __syncthreads();
      if (tid < 128) {
        float m = fmaxf(fmaxf(wmaxs[0][tid], wmaxs[1][tid]),
                        fmaxf(wmaxs[2][tid], wmaxs[3][tid]));
        agg[(size_t)gid * 128 + tid] = m;
      }
    }
  }
}

// ---------------------------------------------------------------------------
// Node-level GEMMs: xnew = x@Wt + bt + relu(agg@Wa + ba)@Wb + bb
// 64 rows per workgroup (4 waves x 16 rows).
// ---------------------------------------------------------------------------
template <int TIN>
__global__ __launch_bounds__(128) void node_mlp(
    const float* __restrict__ xin, const float* __restrict__ agg,
    const float* __restrict__ Wa, const float* __restrict__ ba,
    const float* __restrict__ Wb, const float* __restrict__ bb,
    const float* __restrict__ Wt, const float* __restrict__ bt,
    float* __restrict__ xout) {
  constexpr int TINP = (TIN + 31) & ~31;
  __shared__ alignas(16) _Float16 aT[64 * 128];
  __shared__ alignas(16) _Float16 uT[64 * 128];
  __shared__ alignas(16) _Float16 xT[64 * TINP];
  __shared__ alignas(16) _Float16 wChunkT[128 * 32];
  int tid  = threadIdx.x;
  int lane = tid & 31;
  int wv   = tid >> 5;
  int lh   = lane >> 4;
  int nl   = lane & 15;
  int row0 = blockIdx.x * 64;

  for (int t = tid; t < 64 * 128; t += 128) {
    int r = t >> 7, c = t & 127;
    aT[t] = (_Float16)agg[(size_t)(row0 + r) * 128 + c];
  }
  for (int t = tid; t < 64 * TINP; t += 128) {
    int r = t / TINP, c = t % TINP;
    xT[t] = (_Float16)((c < TIN) ? xin[(size_t)(row0 + r) * TIN + c] : 0.0f);
  }

  // u = relu(agg@Wa + ba)
  v8f acc[8];
#pragma unroll
  for (int nt = 0; nt < 8; ++nt) acc[nt] = (v8f){0.f,0.f,0.f,0.f,0.f,0.f,0.f,0.f};
  gemm_acc<4>(aT, 128, wv * 16, Wa, 128, wChunkT, acc, tid, lane);
#pragma unroll
  for (int nt = 0; nt < 8; ++nt) {
    int n = nt * 16 + nl;
    float bias = ba[n];
#pragma unroll
    for (int r = 0; r < 8; ++r) {
      int row = wv * 16 + r + 8 * lh;
      float v = fmaxf(acc[nt][r] + bias, 0.0f);
      uT[(size_t)row * 128 + n] = (_Float16)v;
    }
  }

  // y = u@Wb ; then += x@Wt ; + bb + bt
  v8f acc2[8];
#pragma unroll
  for (int nt = 0; nt < 8; ++nt) acc2[nt] = (v8f){0.f,0.f,0.f,0.f,0.f,0.f,0.f,0.f};
  gemm_acc<4>(uT, 128, wv * 16, Wb, 128, wChunkT, acc2, tid, lane);
  gemm_acc<TINP / 32>(xT, TINP, wv * 16, Wt, TIN, wChunkT, acc2, tid, lane);
#pragma unroll
  for (int nt = 0; nt < 8; ++nt) {
    int n = nt * 16 + nl;
    float bias = bb[n] + bt[n];
#pragma unroll
    for (int r = 0; r < 8; ++r) {
      int row = row0 + wv * 16 + r + 8 * lh;
      xout[(size_t)row * 128 + n] = acc2[nt][r] + bias;
    }
  }
}

// ---------------------------------------------------------------------------
// Host-side layer driver
// ---------------------------------------------------------------------------
template <int CIN>
static void run_layer(const float* pos, const float* xin, const int* idx,
                      void* const* P, float* agg, float* xout,
                      float* cnt, float* sum1, float* sq1, float* sum2, float* sq2,
                      float* sc1, float* sh1, float* sc2, float* sh2,
                      hipStream_t stream) {
  const float* W1 = (const float*)P[0];  const float* b1  = (const float*)P[1];
  const float* g1 = (const float*)P[2];  const float* be1 = (const float*)P[3];
  const float* W2 = (const float*)P[4];  const float* b2  = (const float*)P[5];
  const float* g2 = (const float*)P[6];  const float* be2 = (const float*)P[7];
  const float* Wa = (const float*)P[8];  const float* ba  = (const float*)P[9];
  const float* Wb = (const float*)P[10]; const float* bb  = (const float*)P[11];
  const float* Wt = (const float*)P[12]; const float* bt  = (const float*)P[13];

  zero_f32<<<dim3(2), dim3(256), 0, stream>>>(sum1, 512);  // sum1,sq1,sum2,sq2
  edge_mlp<0, CIN><<<dim3(NPTS), dim3(128), 0, stream>>>(
      pos, xin, idx, W1, b1, W2, b2, sc1, sh1, sc2, sh2,
      sum1, sq1, sum2, sq2, agg);
  bn_finalize<<<dim3(1), dim3(128), 0, stream>>>(sum1, sq1, cnt, g1, be1, sc1, sh1);
  edge_mlp<1, CIN><<<dim3(NPTS), dim3(128), 0, stream>>>(
      pos, xin, idx, W1, b1, W2, b2, sc1, sh1, sc2, sh2,
      sum1, sq1, sum2, sq2, agg);
  bn_finalize<<<dim3(1), dim3(128), 0, stream>>>(sum2, sq2, cnt, g2, be2, sc2, sh2);
  edge_mlp<2, CIN><<<dim3(NPTS), dim3(128), 0, stream>>>(
      pos, xin, idx, W1, b1, W2, b2, sc1, sh1, sc2, sh2,
      sum1, sq1, sum2, sq2, agg);
  node_mlp<CIN><<<dim3(NPTS / 64), dim3(128), 0, stream>>>(
      xin, agg, Wa, ba, Wb, bb, Wt, bt, xout);
}

extern "C" void kernel_launch(void* const* d_in, const int* in_sizes, int n_in,
                              void* d_out, int out_size, void* d_ws, size_t ws_size,
                              hipStream_t stream) {
  (void)in_sizes; (void)n_in; (void)out_size; (void)ws_size;
  const float* pos  = (const float*)d_in[0];
  const float* feat = (const float*)d_in[1];
  // d_in[2] = batch_indices (implicit via static layout)

  char* ws = (char*)d_ws;
  size_t off = 0;
  int*   idx = (int*)(ws + off);   off += (size_t)NPTS * KK * sizeof(int);
  float* agg = (float*)(ws + off); off += (size_t)NPTS * 128 * sizeof(float);
  float* xA  = (float*)(ws + off); off += (size_t)NPTS * 128 * sizeof(float);
  float* xB  = (float*)(ws + off); off += (size_t)NPTS * 128 * sizeof(float);
  float* st  = (float*)(ws + off);
  float* cnt  = st;            // [1] (region 0..127 reserved)
  float* sum1 = st + 128, *sq1 = st + 256, *sum2 = st + 384, *sq2 = st + 512;
  float* sc1  = st + 640, *sh1 = st + 768, *sc2  = st + 896, *sh2 = st + 1024;

  zero_f32<<<dim3(5), dim3(256), 0, stream>>>(st, 1152);
  knn_kernel<<<dim3(NPTS), dim3(256), 0, stream>>>(pos, idx, cnt);

  void* const* P0 = d_in + 3 + 0 * 14;
  void* const* P1 = d_in + 3 + 1 * 14;
  void* const* P2 = d_in + 3 + 2 * 14;
  run_layer<II>(pos, feat, idx, P0, agg, xA, cnt, sum1, sq1, sum2, sq2,
                sc1, sh1, sc2, sh2, stream);
  run_layer<OO>(pos, xA, idx, P1, agg, xB, cnt, sum1, sq1, sum2, sq2,
                sc1, sh1, sc2, sh2, stream);
  run_layer<OO>(pos, xB, idx, P2, agg, (float*)d_out, cnt, sum1, sq1, sum2, sq2,
                sc1, sh1, sc2, sh2, stream);
}
